// DCRNN_85650237817212
// MI455X (gfx1250) — compile-verified, matrix operand order checked by reading, not run on hardware
//
#include <hip/hip_runtime.h>
#include <hip/hip_bf16.h>

// ---------------- problem constants ----------------
#define NB   200                 // nodes
#define UU   64                  // rnn units
#define BB   64                  // batch
#define TIN  12
#define TOUT 12
#define ROWS (BB * NB)           // 12800 = B*N rows of flattened per-node features

// ---------------- WMMA types ----------------
typedef __attribute__((ext_vector_type(16))) __bf16 bf16x16;
typedef __attribute__((ext_vector_type(8)))  float  f32x8;

// round-to-nearest bias then truncate; packed pair built with one v_perm_b32
__device__ __forceinline__ unsigned int pack2bf(float a, float b) {
  unsigned int ua = __float_as_uint(a) + 0x8000u;
  unsigned int ub = __float_as_uint(b) + 0x8000u;
  // 64-bit {s0=ub, s1=ua}: ua bytes 0..3, ub bytes 4..7
  // out = { ub.b3, ub.b2, ua.b3, ua.b2 }  -> low16 = bf16(a), high16 = bf16(b)
  return __builtin_amdgcn_perm(ub, ua, 0x07060302u);
}
__device__ __forceinline__ unsigned short f2bf(float f) {
  return (unsigned short)((__float_as_uint(f) + 0x8000u) >> 16);
}

// ================= generic bf16-WMMA GEMM, f32 accumulate =================
// C tile 128x64 per block; 256 threads = 8 wave32 waves; wave w owns rows [w*16,w*16+16).
// LDS tiles stored fragment-major: lane's v16bf fragment = two contiguous 16B LDS loads.
// Staging is vectorized: requires K % 8 == 0, N % 4 == 0, lda/ldb % 4 == 0
// (true for every GEMM in this model).
//
// modes: 0: C = alpha*acc + beta*Cin
//        1: gates: g=sigmoid(acc+bias); col<64 -> RH=g*H ; col>=64 -> Ub=g
//        2: cand : c=tanh(acc+bias); C(=h) = u*h + (1-u)*c   (N must be 64)
//        3: C = acc + bias
#define TM 128
#define TN 64
#define TKK 32
#define NTHR 256

__global__ __launch_bounds__(NTHR)
void wmma_gemm_kernel(const float* __restrict__ A, int lda, long long sAz,
                      const float* __restrict__ Bm, int ldb, long long sBz,
                      const float* __restrict__ Cin, int ldcin, long long sCinz,
                      float* __restrict__ C, int ldc, long long sCz,
                      const float* __restrict__ bias,
                      const float* __restrict__ Hbuf,
                      const float* __restrict__ Ubuf,
                      float* __restrict__ RHbuf,
                      int M, int N, int K,
                      float alpha, float beta, int mode) {
  __shared__ __align__(16) unsigned short As[TM * TKK];   // 8 KB, fragment-major
  __shared__ __align__(16) unsigned short Bs[TKK * TN];   // 4 KB, fragment-major

  const int z = blockIdx.z;
  A  += (long long)z * sAz;
  Bm += (long long)z * sBz;
  C  += (long long)z * sCz;
  if (Cin) Cin += (long long)z * sCinz;

  const int tid  = threadIdx.x;
  const int lane = tid & 31;            // wave32
  const int w    = tid >> 5;            // wave id 0..7
  const int half = lane >> 4;
  const int nl   = lane & 15;

  const int rowBase = blockIdx.y * TM;
  const int colBase = blockIdx.x * TN;

  // ---- precompute A staging tasks: 512 8-wide tasks, 2 per thread ----
  // task t: row i = t>>2 ; sub = t&3 -> (g = sub>>1, hf = sub&1); kk0 = g*16+hf*8
  int          aKk0[2];
  uint4*       aDst[2];
  const float* aRow[2];
  bool         aOK[2];
#pragma unroll
  for (int q = 0; q < 2; ++q) {
    int t  = tid + q * NTHR;
    int i  = t >> 2, sub = t & 3;
    int hf = sub & 1, g = sub >> 1;
    aKk0[q] = g * 16 + hf * 8;
    int slot = (i >> 4) * 32 + hf * 16 + (i & 15);
    aDst[q] = (uint4*)(As + slot * 16 + g * 8);
    int gr  = rowBase + i;
    aOK[q]  = (gr < M);
    aRow[q] = A + (long long)gr * lda;
  }
  // ---- precompute B staging tasks: 512 4-wide tasks, 2 per thread ----
  // task t: kk = t>>4 ; ng = t&15 -> cols [colBase + ng*4, +4)
  int             bKk[2];
  unsigned short* bDst[2];
  const float*    bCol[2];
  bool            bOK[2];
#pragma unroll
  for (int q = 0; q < 2; ++q) {
    int t  = tid + q * NTHR;
    int kk = t >> 4, ng = t & 15;
    int hf = (kk >> 3) & 1, g = kk >> 4, r = kk & 7;
    int slot0 = (ng >> 2) * 32 + hf * 16 + (ng & 3) * 4;
    bDst[q] = Bs + slot0 * 16 + g * 8 + r;
    bKk[q]  = kk;
    int gc  = colBase + ng * 4;
    bOK[q]  = (gc < N);
    bCol[q] = Bm + gc;
  }

  f32x8 acc0, acc1, acc2, acc3;
#pragma unroll
  for (int i = 0; i < 8; ++i) { acc0[i] = 0.f; acc1[i] = 0.f; acc2[i] = 0.f; acc3[i] = 0.f; }

  for (int k0 = 0; k0 < K; k0 += TKK) {
    // ---- stage A tile: 2 x (2 b128 global loads -> 1 b128 LDS store) ----
#pragma unroll
    for (int q = 0; q < 2; ++q) {
      float4 f0 = make_float4(0.f, 0.f, 0.f, 0.f);
      float4 f1 = make_float4(0.f, 0.f, 0.f, 0.f);
      int gk = k0 + aKk0[q];
      if (aOK[q] && gk < K) {
        const float4* p = (const float4*)(aRow[q] + gk);
        f0 = p[0]; f1 = p[1];
      }
      uint4 d;
      d.x = pack2bf(f0.x, f0.y);
      d.y = pack2bf(f0.z, f0.w);
      d.z = pack2bf(f1.x, f1.y);
      d.w = pack2bf(f1.z, f1.w);
      *aDst[q] = d;
    }
    // ---- stage B tile: 2 x (1 b128 global load -> 4 b16 LDS stores) ----
#pragma unroll
    for (int q = 0; q < 2; ++q) {
      float4 f = make_float4(0.f, 0.f, 0.f, 0.f);
      int gk = k0 + bKk[q];
      if (bOK[q] && gk < K) f = *(const float4*)(bCol[q] + (long long)gk * ldb);
      unsigned short* d = bDst[q];
      d[0]  = f2bf(f.x);
      d[16] = f2bf(f.y);
      d[32] = f2bf(f.z);
      d[48] = f2bf(f.w);
    }
    __syncthreads();

    union V16 { uint4 q[2]; bf16x16 v; };
    V16 af, b0, b1, b2, b3;
    const uint4* ap = (const uint4*)(As + (w * 32 + lane) * 16);
    af.q[0] = ap[0]; af.q[1] = ap[1];
    const uint4* bp0 = (const uint4*)(Bs + (lane) * 16);
    const uint4* bp1 = (const uint4*)(Bs + (32 + lane) * 16);
    const uint4* bp2 = (const uint4*)(Bs + (64 + lane) * 16);
    const uint4* bp3 = (const uint4*)(Bs + (96 + lane) * 16);
    b0.q[0] = bp0[0]; b0.q[1] = bp0[1];
    b1.q[0] = bp1[0]; b1.q[1] = bp1[1];
    b2.q[0] = bp2[0]; b2.q[1] = bp2[1];
    b3.q[0] = bp3[0]; b3.q[1] = bp3[1];
    __syncthreads();   // fragments in registers; LDS free for next stage

    acc0 = __builtin_amdgcn_wmma_f32_16x16x32_bf16(false, af.v, false, b0.v, (short)0, acc0, false, false);
    acc1 = __builtin_amdgcn_wmma_f32_16x16x32_bf16(false, af.v, false, b1.v, (short)0, acc1, false, false);
    acc2 = __builtin_amdgcn_wmma_f32_16x16x32_bf16(false, af.v, false, b2.v, (short)0, acc2, false, false);
    acc3 = __builtin_amdgcn_wmma_f32_16x16x32_bf16(false, af.v, false, b3.v, (short)0, acc3, false, false);
  }

  // ---- epilogue: C/D layout -> lane(n=lane%16), row = half*8 + vgpr_idx ----
#pragma unroll
  for (int cb = 0; cb < 4; ++cb) {
    f32x8 a = (cb == 0) ? acc0 : (cb == 1) ? acc1 : (cb == 2) ? acc2 : acc3;
#pragma unroll
    for (int i = 0; i < 8; ++i) {
      int gr = rowBase + w * 16 + half * 8 + i;
      int gc = colBase + cb * 16 + nl;
      if (gr >= M || gc >= N) continue;
      float v = a[i];
      if (mode == 0) {
        v *= alpha;
        if (Cin) v += beta * Cin[(long long)gr * ldcin + gc];
        C[(long long)gr * ldc + gc] = v;
      } else if (mode == 1) {
        float g = 1.f / (1.f + __expf(-(v + bias[gc])));
        long long rr = (long long)gr * UU;
        if (gc < UU) RHbuf[rr + gc] = g * Hbuf[rr + gc];
        else         ((float*)Ubuf)[rr + (gc - UU)] = g;
      } else if (mode == 2) {
        float cnd = tanhf(v + bias[gc]);
        long long idx = (long long)gr * UU + gc;
        float u = Ubuf[idx], h = Hbuf[idx];
        C[idx] = u * h + (1.f - u) * cnd;
      } else {
        C[(long long)gr * ldc + gc] = v + bias[gc];
      }
    }
  }
}

// ================= small helper kernels =================
__global__ void pack_x_kernel(float* __restrict__ Xbuf, int ldx,
                              const float* __restrict__ X, long long bstride, int D) {
  int i = blockIdx.x * blockDim.x + threadIdx.x;
  int total = ROWS * D;
  if (i >= total) return;
  int r = i / D, c = i - r * D;
  int b = r / NB, n = r - b * NB;
  Xbuf[(long long)r * ldx + c] = X[(long long)b * bstride + (long long)n * D + c];
}

__global__ void pack_state_kernel(float* __restrict__ Xbuf, int ldx, int D,
                                  const float* __restrict__ H) {
  int i = blockIdx.x * blockDim.x + threadIdx.x;
  if (i >= ROWS * UU) return;
  int r = i >> 6, c = i & 63;
  Xbuf[(long long)r * ldx + D + c] = H[i];
}

// Wperm[k*F + f, :] = W[f*3 + k, :]  (Chebyshev-interleaved -> block layout)
__global__ void permute_w_kernel(float* __restrict__ out, const float* __restrict__ in,
                                 int F, int nc) {
  int i = blockIdx.x * blockDim.x + threadIdx.x;
  int total = 3 * F * nc;
  if (i >= total) return;
  int j = i % nc, row = i / nc;
  int k = row / F, f = row - k * F;
  out[i] = in[(long long)(f * 3 + k) * nc + j];
}

__global__ void zero_kernel(float* p, int n) {
  int i = blockIdx.x * blockDim.x + threadIdx.x;
  if (i < n) p[i] = 0.f;
}

__global__ void scatter_out_kernel(float* __restrict__ out, const float* __restrict__ proj, int t) {
  int i = blockIdx.x * blockDim.x + threadIdx.x;
  int total = BB * NB * NB;
  if (i >= total) return;
  int b = i / (NB * NB), rest = i - b * (NB * NB);
  out[(long long)b * TOUT * (NB * NB) + (long long)t * (NB * NB) + rest] = proj[i];
}

// ================= host orchestration =================
extern "C" void kernel_launch(void* const* d_in, const int* in_sizes, int n_in,
                              void* d_out, int out_size, void* d_ws, size_t ws_size,
                              hipStream_t stream) {
  (void)in_sizes; (void)n_in; (void)out_size; (void)ws_size;
  const float* inputs  = (const float*)d_in[0];
  const float* support = (const float*)d_in[1];
  const float* eWg0 = (const float*)d_in[2];  const float* ebg0 = (const float*)d_in[3];
  const float* eWc0 = (const float*)d_in[4];  const float* ebc0 = (const float*)d_in[5];
  const float* eWg  = (const float*)d_in[6];  const float* ebg  = (const float*)d_in[7];
  const float* eWc  = (const float*)d_in[8];  const float* ebc  = (const float*)d_in[9];
  const float* dWg0 = (const float*)d_in[10]; const float* dbg0 = (const float*)d_in[11];
  const float* dWc0 = (const float*)d_in[12]; const float* dbc0 = (const float*)d_in[13];
  const float* dWg  = (const float*)d_in[14]; const float* dbg  = (const float*)d_in[15];
  const float* dWc  = (const float*)d_in[16]; const float* dbc  = (const float*)d_in[17];
  const float* projW = (const float*)d_in[18]; const float* projb = (const float*)d_in[19];
  float* out = (float*)d_out;

  // ---- workspace carve (all f32) ----
  float* ws = (float*)d_ws;
  size_t o = 0;
  float* Xbuf = ws + o; o += (size_t)ROWS * 792;        // (12800, 3F) max F0*3=792
  float* h    = ws + o; o += (size_t)4 * ROWS * UU;     // 4 layer states
  float* rh   = ws + o; o += (size_t)ROWS * UU;
  float* ub   = ws + o; o += (size_t)ROWS * UU;
  float* proj = ws + o; o += (size_t)ROWS * NB;         // (12800, 200) = (B, N*N)
  float* eWg0p = ws + o; o += (size_t)792 * 128;
  float* eWc0p = ws + o; o += (size_t)792 * 64;
  float* eWgp  = ws + o; o += (size_t)3 * 384 * 128;
  float* eWcp  = ws + o; o += (size_t)3 * 384 * 64;
  float* dWg0p = ws + o; o += (size_t)792 * 128;
  float* dWc0p = ws + o; o += (size_t)792 * 64;
  float* dWgp  = ws + o; o += (size_t)3 * 384 * 128;
  float* dWcp  = ws + o; o += (size_t)3 * 384 * 64;

  auto gemm = [&](const float* A, int lda, long long sAz,
                  const float* Bm, int ldb, long long sBz,
                  const float* Cin, int ldcin, long long sCinz,
                  float* C, int ldc, long long sCz,
                  const float* bias, const float* H, const float* U_, float* RH,
                  int M, int N, int K, float alpha, float beta, int mode, int Z) {
    dim3 g((N + TN - 1) / TN, (M + TM - 1) / TM, Z), b(NTHR);
    wmma_gemm_kernel<<<g, b, 0, stream>>>(A, lda, sAz, Bm, ldb, sBz, Cin, ldcin, sCinz,
                                          C, ldc, sCz, bias, H, U_, RH,
                                          M, N, K, alpha, beta, mode);
  };

  // ---- one-time (per launch, deterministic) setup ----
  {
    int n1 = 4 * ROWS * UU;
    zero_kernel<<<(n1 + 255) / 256, 256, 0, stream>>>(h, n1);
    int n2 = ROWS * NB;
    zero_kernel<<<(n2 + 255) / 256, 256, 0, stream>>>(proj, n2);
    auto perm = [&](float* dst, const float* src, int F, int nc) {
      int tot = 3 * F * nc;
      permute_w_kernel<<<(tot + 255) / 256, 256, 0, stream>>>(dst, src, F, nc);
    };
    perm(eWg0p, eWg0, 264, 128); perm(eWc0p, eWc0, 264, 64);
    perm(dWg0p, dWg0, 264, 128); perm(dWc0p, dWc0, 264, 64);
    for (int l = 0; l < 3; ++l) {
      perm(eWgp + (size_t)l * 384 * 128, eWg + (size_t)l * 384 * 128, 128, 128);
      perm(eWcp + (size_t)l * 384 * 64,  eWc + (size_t)l * 384 * 64,  128, 64);
      perm(dWgp + (size_t)l * 384 * 128, dWg + (size_t)l * 384 * 128, 128, 128);
      perm(dWcp + (size_t)l * 384 * 64,  dWc + (size_t)l * 384 * 64,  128, 64);
    }
  }

  // ---- one DCGRU cell ----
  auto run_cell = [&](const float* x, long long xbs, int D, float* hl,
                      const float* Wgp, const float* bg,
                      const float* Wcp, const float* bc) {
    int F = D + UU, K3 = 3 * F;
    long long sZ = (long long)NB * K3;                // per-batch row stride in Xbuf
    int totx = ROWS * D;
    pack_x_kernel<<<(totx + 255) / 256, 256, 0, stream>>>(Xbuf, K3, x, xbs, D);
    pack_state_kernel<<<(ROWS * UU + 255) / 256, 256, 0, stream>>>(Xbuf, K3, D, hl);
    // X1 = S @ X0   (batched over b via grid.z)
    gemm(support, NB, 0, Xbuf, K3, sZ, nullptr, 0, 0, Xbuf + F, K3, sZ,
         nullptr, nullptr, nullptr, nullptr, NB, F, NB, 1.f, 0.f, 0, BB);
    // X2 = 2*S@X1 - X0
    gemm(support, NB, 0, Xbuf + F, K3, sZ, Xbuf, K3, sZ, Xbuf + 2 * F, K3, sZ,
         nullptr, nullptr, nullptr, nullptr, NB, F, NB, 2.f, -1.f, 0, BB);
    // gates: sigmoid([X0|X1|X2] @ Wg + bg) -> rh = r*h, ub = u
    gemm(Xbuf, K3, 0, Wgp, 2 * UU, 0, nullptr, 0, 0, rh, UU, 0,
         bg, hl, ub, rh, ROWS, 2 * UU, K3, 1.f, 0.f, 1, 1);
    // rebuild state columns with r*h, redo diffusion
    pack_state_kernel<<<(ROWS * UU + 255) / 256, 256, 0, stream>>>(Xbuf, K3, D, rh);
    gemm(support, NB, 0, Xbuf, K3, sZ, nullptr, 0, 0, Xbuf + F, K3, sZ,
         nullptr, nullptr, nullptr, nullptr, NB, F, NB, 1.f, 0.f, 0, BB);
    gemm(support, NB, 0, Xbuf + F, K3, sZ, Xbuf, K3, sZ, Xbuf + 2 * F, K3, sZ,
         nullptr, nullptr, nullptr, nullptr, NB, F, NB, 2.f, -1.f, 0, BB);
    // candidate + GRU update fused: h = u*h + (1-u)*tanh(...)
    gemm(Xbuf, K3, 0, Wcp, UU, 0, nullptr, 0, 0, hl, UU, 0,
         bc, hl, ub, nullptr, ROWS, UU, K3, 1.f, 0.f, 2, 1);
  };

  // ---- encoder ----
  for (int t = 0; t < TIN; ++t) {
    run_cell(inputs + (long long)t * NB * NB, (long long)TIN * NB * NB, NB,
             h, eWg0p, ebg0, eWc0p, ebc0);
    for (int l = 1; l < 4; ++l)
      run_cell(h + (size_t)(l - 1) * ROWS * UU, (long long)NB * UU, UU,
               h + (size_t)l * ROWS * UU,
               eWgp + (size_t)(l - 1) * 384 * 128, ebg + (size_t)(l - 1) * 128,
               eWcp + (size_t)(l - 1) * 384 * 64,  ebc + (size_t)(l - 1) * 64);
  }

  // ---- decoder ----
  for (int t = 0; t < TOUT; ++t) {
    run_cell(proj, (long long)NB * NB, NB, h, dWg0p, dbg0, dWc0p, dbc0);
    for (int l = 1; l < 4; ++l)
      run_cell(h + (size_t)(l - 1) * ROWS * UU, (long long)NB * UU, UU,
               h + (size_t)l * ROWS * UU,
               dWgp + (size_t)(l - 1) * 384 * 128, dbg + (size_t)(l - 1) * 128,
               dWcp + (size_t)(l - 1) * 384 * 64,  dbc + (size_t)(l - 1) * 64);
    // projection: (12800,64) @ (64,200) + b -> proj, then scatter into d_out[:, t, :]
    gemm(h + (size_t)3 * ROWS * UU, UU, 0, projW, NB, 0, nullptr, 0, 0, proj, NB, 0,
         projb, nullptr, nullptr, nullptr, ROWS, NB, UU, 1.f, 0.f, 3, 1);
    int tot = BB * NB * NB;
    scatter_out_kernel<<<(tot + 255) / 256, 256, 0, stream>>>(out, proj, t);
  }
}
